// MultiDismantler_net_4355096838398
// MI455X (gfx1250) — matrix operand order repaired and checked by reading, not compiled, not run on hardware
//
#include <hip/hip_runtime.h>
#include <cstdint>

#define EMB 128
#define NNODES 100000
#define MVIRT 64
#define NNZE 1600000
#define NLAYERS 2
#define BPITERS 3

typedef __attribute__((ext_vector_type(16))) __bf16 v16bf;
typedef __attribute__((ext_vector_type(8)))  __bf16 v8bf;
typedef __attribute__((ext_vector_type(4)))  __bf16 v4bf;
typedef __attribute__((ext_vector_type(8)))  float  v8f;

#if __has_builtin(__builtin_amdgcn_global_load_async_to_lds_b128)
#define HAVE_ASYNC_LDS 1
// The builtin takes pointers to 16-byte int vectors in addrspace(1)/(3).
typedef int v4i_vec __attribute__((vector_size(16)));
typedef __attribute__((address_space(1))) v4i_vec gv4i;
typedef __attribute__((address_space(3))) v4i_vec lv4i;
#define AS_GLOBAL(p) ((gv4i*)(uintptr_t)(p))
#define AS_LDS(p)    ((lv4i*)(uint32_t)(uintptr_t)(p))
__device__ __forceinline__ void wait_async_zero() {
#if __has_builtin(__builtin_amdgcn_s_wait_asynccnt)
  __builtin_amdgcn_s_wait_asynccnt(0);
#else
  asm volatile("s_wait_asynccnt 0x0" ::: "memory");
#endif
}
#else
#define HAVE_ASYNC_LDS 0
#endif

// ---------------------------------------------------------------------------
// prep: q1t[n][k] = (p1 @ p3_top)[k][n], q2t[n][k] = (p2 @ p3_bot)[k][n]
// stored TRANSPOSED in bf16 so B-matrix WMMA fragments are contiguous loads.
// grid: 128 blocks (n), 128 threads (k)
// ---------------------------------------------------------------------------
__global__ void prep_q_kernel(const float* __restrict__ p1,
                              const float* __restrict__ p2,
                              const float* __restrict__ p3,
                              __bf16* __restrict__ q1t,
                              __bf16* __restrict__ q2t) {
  const int n = blockIdx.x;
  const int k = threadIdx.x;
  float s1 = 0.f, s2 = 0.f;
  for (int j = 0; j < EMB; ++j) {
    s1 = fmaf(p1[k * EMB + j], p3[j * EMB + n], s1);
    s2 = fmaf(p2[k * EMB + j], p3[(EMB + j) * EMB + n], s2);
  }
  q1t[n * EMB + k] = (__bf16)s1;
  q2t[n * EMB + k] = (__bf16)s2;
}

// y0 = l2norm(relu(w_n2l[0] + w_n2l[1])) broadcast to 64 rows (bf16). 128 thr.
__global__ void prep_y0_kernel(const float* __restrict__ w_n2l,
                               __bf16* __restrict__ y0) {
  __shared__ float red[EMB];
  const int c = threadIdx.x;
  float v = fmaxf(w_n2l[c] + w_n2l[EMB + c], 0.f);
  red[c] = v * v;
  __syncthreads();
  for (int off = 64; off; off >>= 1) {
    if (c < off) red[c] += red[c + off];
    __syncthreads();
  }
  const float inv = 1.f / fmaxf(sqrtf(red[0]), 1e-12f);
  const __bf16 bv = (__bf16)(v * inv);
  for (int r = 0; r < MVIRT; ++r) y0[r * EMB + c] = bv;
}

// enc = l2norm(node_input_last @ W_enc + b_enc), bf16.  One wave per node.
__global__ __launch_bounds__(256)
void encode_kernel(const float* __restrict__ x,      // [N][3]
                   const float* __restrict__ W,      // [3][128]
                   const float* __restrict__ b,      // [128]
                   __bf16* __restrict__ enc, int n) {
  const int lane = threadIdx.x & 31;
  const int w = (blockIdx.x * blockDim.x + threadIdx.x) >> 5;
  if (w >= n) return;
  const float x0 = x[(size_t)w * 3 + 0];
  const float x1 = x[(size_t)w * 3 + 1];
  const float x2 = x[(size_t)w * 3 + 2];
  float e[4];
  float ss = 0.f;
#pragma unroll
  for (int s = 0; s < 4; ++s) {
    const int c = lane + 32 * s;
    e[s] = fmaf(x0, W[c], fmaf(x1, W[EMB + c], fmaf(x2, W[2 * EMB + c], b[c])));
    ss = fmaf(e[s], e[s], ss);
  }
#pragma unroll
  for (int off = 16; off; off >>= 1) ss += __shfl_xor(ss, off, 32);
  const float inv = 1.f / fmaxf(sqrtf(ss), 1e-12f);
#pragma unroll
  for (int s = 0; s < 4; ++s)
    enc[(size_t)w * EMB + lane + 32 * s] = (__bf16)(e[s] * inv);
}

// pool[row[e]] += val[e] * cur[col[e]]   (one wave per edge, grid-stride).
// cur in bf16 (256B/row gather, L2-resident); pool accumulated fp32 atomics.
__global__ __launch_bounds__(256)
void spmm_edges_kernel(const int* __restrict__ row, const int* __restrict__ col,
                       const float* __restrict__ val,
                       const __bf16* __restrict__ cur,
                       float* __restrict__ pool, int nnz) {
  const int lane = threadIdx.x & 31;
  const int w0 = (blockIdx.x * blockDim.x + threadIdx.x) >> 5;
  const int nw = (gridDim.x * blockDim.x) >> 5;
  for (int e = w0; e < nnz; e += nw) {
    // prefetch next iteration's gather row (everything is L2-resident)
    const int e2 = e + nw;
    if (e2 < nnz) {
      const int c2 = col[e2];
      __builtin_prefetch(cur + (size_t)c2 * EMB + lane * 4, 0, 1);
    }
    const int r = row[e];
    const int c = col[e];
    const float v = val[e];
    const v4bf d = *(const v4bf*)(cur + (size_t)c * EMB + lane * 4);
    float* dst = pool + (size_t)r * EMB + lane * 4;
    atomicAdd(dst + 0, v * (float)d[0]);
    atomicAdd(dst + 1, v * (float)d[1]);
    atomicAdd(dst + 2, v * (float)d[2]);
    atomicAdd(dst + 3, v * (float)d[3]);
  }
}

// ypool[srow[i]] += sval[i] * cur[scol[i]]  with 64x128 LDS accumulator/block.
__global__ __launch_bounds__(256)
void spmm_subg_kernel(const int* __restrict__ srow, const int* __restrict__ scol,
                      const float* __restrict__ sval,
                      const __bf16* __restrict__ cur,
                      float* __restrict__ ypool, int n) {
  __shared__ float acc[MVIRT * EMB];  // 32 KB
  for (int i = threadIdx.x; i < MVIRT * EMB; i += 256) acc[i] = 0.f;
  __syncthreads();
  const int half = threadIdx.x >> 7;  // two nodes per pass
  const int c = threadIdx.x & 127;
  for (int base = blockIdx.x * 2; base < n; base += gridDim.x * 2) {
    const int node = base + half;
    if (node < n) {
      const int r = srow[node];
      const int cc = scol[node];
      const float v = sval[node];
      atomicAdd(&acc[r * EMB + c], v * (float)cur[(size_t)cc * EMB + c]);
    }
  }
  __syncthreads();
  for (int i = threadIdx.x; i < MVIRT * EMB; i += 256)
    atomicAdd(&ypool[i], acc[i]);
}

// ---------------------------------------------------------------------------
// cur' = l2norm(relu(pool @ q1 + cur @ q2))   via v_wmma_f32_16x16x32_bf16.
// 8 waves/block; wave owns a 16-col output tile; q fragments persist in VGPRs.
// A fragments are preloaded so the 8 WMMAs issue back-to-back on two
// independent accumulator chains.
// ---------------------------------------------------------------------------
__device__ __forceinline__ v16bf load_a_frag(const __bf16* rowp, int ks, int h) {
  // A 16x32 bf16, contiguous k order used here (matches B): lane (m + 16h), elems 0..15 -> K = 32ks+16h+e.
  const __bf16* p = rowp + ks * 32 + 16 * h;
  const v8bf lo = *(const v8bf*)(p);
  const v8bf hi = *(const v8bf*)(p + 8);
  return __builtin_shufflevector(lo, hi, 0, 1, 2, 3, 4, 5, 6, 7,
                                 8, 9, 10, 11, 12, 13, 14, 15);
}

__global__ __launch_bounds__(256)
void update_kernel(const float* __restrict__ pool,
                   const __bf16* __restrict__ curin,
                   const __bf16* __restrict__ q1t,   // [n][k] transposed bf16
                   const __bf16* __restrict__ q2t,
                   __bf16* __restrict__ curout,
                   float* __restrict__ outf,         // null except last iter
                   int rows) {
  __shared__ __bf16 sA0[16][EMB];   // pool tile (bf16)    4 KB
  __shared__ __bf16 sA1[16][EMB];   // cur  tile           4 KB
  __shared__ float  sZ[16][EMB];    // pre-norm output     8 KB
  __shared__ float  sNorm[16];

  const int tid = threadIdx.x;
  const int lane = tid & 31;
  const int wave = tid >> 5;   // column tile 0..7
  const int m = lane & 15;
  const int h = lane >> 4;

  // Persistent B fragments. B 32x16 bf16, contiguous k order used here (matches A): lane (n + 16h) holds
  // K rows 16h..16h+15 for column n; elem e -> K = 32ks + 16h + e.
  v16bf bq1[4], bq2[4];
  {
    const int n = wave * 16 + m;
#pragma unroll
    for (int ks = 0; ks < 4; ++ks) {
      bq1[ks] = *(const v16bf*)(q1t + n * EMB + ks * 32 + 16 * h);
      bq2[ks] = *(const v16bf*)(q2t + n * EMB + ks * 32 + 16 * h);
    }
  }

  const int ntiles = rows >> 4;
  for (int rt = blockIdx.x; rt < ntiles; rt += gridDim.x) {
    const size_t row0 = (size_t)rt * 16;
    __syncthreads();
    {  // stage A tiles into LDS (pool: f32 -> bf16; cur: async global->LDS)
#if HAVE_ASYNC_LDS
      __builtin_amdgcn_global_load_async_to_lds_b128(
          AS_GLOBAL(curin + row0 * EMB + tid * 8),
          AS_LDS(&sA1[0][0] + tid * 8), 0, 0);
#else
      ((uint4*)sA1)[tid] = ((const uint4*)(curin + row0 * EMB))[tid];
#endif
      const int base = tid * 8;
      const int mm = base >> 7;
      const int kk = base & 127;
      const float4* ps = (const float4*)(pool + (row0 + mm) * EMB + kk);
      const float4 f0 = ps[0], f1 = ps[1];
      v8bf pv;
      pv[0] = (__bf16)f0.x; pv[1] = (__bf16)f0.y;
      pv[2] = (__bf16)f0.z; pv[3] = (__bf16)f0.w;
      pv[4] = (__bf16)f1.x; pv[5] = (__bf16)f1.y;
      pv[6] = (__bf16)f1.z; pv[7] = (__bf16)f1.w;
      *(v8bf*)(&sA0[mm][kk]) = pv;
      if (tid < 16) sNorm[tid] = 0.f;
#if HAVE_ASYNC_LDS
      wait_async_zero();
#endif
    }
    __syncthreads();

    // Preload all A fragments, then issue WMMAs back-to-back on two chains.
    v16bf aP[4], aC[4];
#pragma unroll
    for (int ks = 0; ks < 4; ++ks) {
      aP[ks] = load_a_frag(&sA0[m][0], ks, h);
      aC[ks] = load_a_frag(&sA1[m][0], ks, h);
    }
    v8f acc1 = {}, acc2 = {};
#pragma unroll
    for (int ks = 0; ks < 4; ++ks) {
      acc1 = __builtin_amdgcn_wmma_f32_16x16x32_bf16(
          false, aP[ks], false, bq1[ks], (short)0, acc1, false, false);
      acc2 = __builtin_amdgcn_wmma_f32_16x16x32_bf16(
          false, aC[ks], false, bq2[ks], (short)0, acc2, false, false);
    }
    // relu + scatter into LDS (C/D layout: elem r, lane -> M=r+8h, N=lane&15)
#pragma unroll
    for (int r = 0; r < 8; ++r)
      sZ[r + 8 * h][wave * 16 + m] = fmaxf(acc1[r] + acc2[r], 0.f);
    __syncthreads();

    if (tid < 128) {  // row squared-norms: 8 threads x 16 cols per row
      const int rr = tid >> 3;
      const int c0 = (tid & 7) * 16;
      float s = 0.f;
#pragma unroll
      for (int i = 0; i < 16; ++i) {
        const float z = sZ[rr][c0 + i];
        s = fmaf(z, z, s);
      }
      atomicAdd(&sNorm[rr], s);
    }
    __syncthreads();

    {  // normalize + store bf16 (and optional fp32 final output)
      const int base = tid * 8;
      const int mm = base >> 7;
      const int kk = base & 127;
      const float inv = 1.f / fmaxf(sqrtf(sNorm[mm]), 1e-12f);
      v8bf ov;
#pragma unroll
      for (int i = 0; i < 8; ++i) {
        const float wv = sZ[mm][kk + i] * inv;
        ov[i] = (__bf16)wv;
        if (outf) outf[(row0 + mm) * EMB + kk + i] = wv;
      }
      *(v8bf*)(&curout[(row0 + mm) * EMB + kk]) = ov;
    }
  }
}

// ---------------------------------------------------------------------------
extern "C" void kernel_launch(void* const* d_in, const int* in_sizes, int n_in,
                              void* d_out, int out_size, void* d_ws,
                              size_t ws_size, hipStream_t stream) {
  (void)in_sizes; (void)n_in; (void)out_size; (void)ws_size;
  const float* node_input = (const float*)d_in[0];
  const float* W_enc      = (const float*)d_in[1];
  const float* b_enc      = (const float*)d_in[2];
  const float* w_n2l      = (const float*)d_in[3];
  const float* p1         = (const float*)d_in[4];
  const float* p2         = (const float*)d_in[5];
  const float* p3         = (const float*)d_in[6];
  const int*   n2n_row    = (const int*)d_in[7];
  const int*   n2n_col    = (const int*)d_in[8];
  const float* n2n_val    = (const float*)d_in[9];
  const int*   subg_row   = (const int*)d_in[10];
  const int*   subg_col   = (const int*)d_in[11];
  const float* subg_val   = (const float*)d_in[12];
  float* out = (float*)d_out;

  char* ws = (char*)d_ws;
  size_t off = 0;
  auto alloc = [&](size_t bytes) -> void* {
    void* p = ws + off;
    off = (off + bytes + 255) & ~(size_t)255;
    return p;
  };
  __bf16* q1t  = (__bf16*)alloc((size_t)EMB * EMB * sizeof(__bf16));
  __bf16* q2t  = (__bf16*)alloc((size_t)EMB * EMB * sizeof(__bf16));
  __bf16* y0   = (__bf16*)alloc((size_t)MVIRT * EMB * sizeof(__bf16));
  __bf16* yA   = (__bf16*)alloc((size_t)MVIRT * EMB * sizeof(__bf16));
  __bf16* yB   = (__bf16*)alloc((size_t)MVIRT * EMB * sizeof(__bf16));
  float*  ypool = (float*)alloc((size_t)MVIRT * EMB * sizeof(float));
  __bf16* enc  = (__bf16*)alloc((size_t)NNODES * EMB * sizeof(__bf16));
  __bf16* curA = (__bf16*)alloc((size_t)NNODES * EMB * sizeof(__bf16));
  __bf16* curB = (__bf16*)alloc((size_t)NNODES * EMB * sizeof(__bf16));
  float*  pool = (float*)alloc((size_t)NNODES * EMB * sizeof(float));

  prep_q_kernel<<<EMB, EMB, 0, stream>>>(p1, p2, p3, q1t, q2t);
  prep_y0_kernel<<<1, EMB, 0, stream>>>(w_n2l, y0);
  // both layers consume encoding of node_input[LAYERS-1] (reference quirk)
  encode_kernel<<<(NNODES * 32 + 255) / 256, 256, 0, stream>>>(
      node_input + (size_t)(NLAYERS - 1) * NNODES * 3, W_enc, b_enc, enc,
      NNODES);

  __bf16* couts[BPITERS] = {curA, curB, curA};
  __bf16* youts[BPITERS] = {yA, yB, yA};
  for (int l = 0; l < NLAYERS; ++l) {
    const __bf16* cin = enc;
    const __bf16* yin = y0;
    for (int it = 0; it < BPITERS; ++it) {
      (void)hipMemsetAsync(pool, 0, (size_t)NNODES * EMB * sizeof(float),
                           stream);
      (void)hipMemsetAsync(ypool, 0, (size_t)MVIRT * EMB * sizeof(float),
                           stream);
      spmm_edges_kernel<<<8192, 256, 0, stream>>>(
          n2n_row + (size_t)l * NNZE, n2n_col + (size_t)l * NNZE,
          n2n_val + (size_t)l * NNZE, cin, pool, NNZE);
      spmm_subg_kernel<<<768, 256, 0, stream>>>(
          subg_row + (size_t)l * NNODES, subg_col + (size_t)l * NNODES,
          subg_val + (size_t)l * NNODES, cin, ypool, NNODES);
      float* ofn = (it == BPITERS - 1)
                       ? out + (size_t)l * (NNODES + MVIRT) * EMB : nullptr;
      float* ofy = (it == BPITERS - 1)
                       ? out + ((size_t)l * (NNODES + MVIRT) + NNODES) * EMB
                       : nullptr;
      update_kernel<<<1024, 256, 0, stream>>>(pool, cin, q1t, q2t, couts[it],
                                              ofn, NNODES);
      update_kernel<<<4, 256, 0, stream>>>(ypool, yin, q1t, q2t, youts[it],
                                           ofy, MVIRT);
      cin = couts[it];
      yin = youts[it];
    }
  }
}
